// MoEGate_65816078844550
// MI455X (gfx1250) — matrix operand order repaired and checked by reading, not compiled, not run on hardware
//
#include <hip/hip_runtime.h>
#include <hip/hip_bf16.h>

typedef __attribute__((ext_vector_type(16))) __bf16 v16bf;
typedef __attribute__((ext_vector_type(8)))  float  v8f;

namespace {
constexpr int kTok    = 16384;   // 4 * 4096 tokens
constexpr int kHid    = 2048;
constexpr int kExp    = 160;
constexpr int kTiles  = 10;      // 160 / 16 expert tiles
constexpr int kChunks = 64;      // 2048 / 32 K-chunks
constexpr int kTopK   = 6;
constexpr int kGroups = 8;
constexpr int kEpg    = 20;      // experts per group
constexpr int kTopG   = 3;
constexpr int kTPB    = 128;     // 4 waves
constexpr int kTokPW  = 32;      // 32 tokens per wave (2 M-tiles, B reused 2x)
constexpr int kTokPB  = 128;     // tokens per block
}

union BV { uint4 u[2]; v16bf v; };

__global__ void zero_accum(float* __restrict__ p, int n) {
  int i = blockIdx.x * blockDim.x + threadIdx.x;
  if (i < n) p[i] = 0.0f;
}

__global__ void wconv(const float* __restrict__ w, __bf16* __restrict__ wbf, int n) {
  int i = blockIdx.x * blockDim.x + threadIdx.x;
  if (i < n) wbf[i] = (__bf16)w[i];
}

__device__ __forceinline__ v16bf cvt_a_frag(const float* xrow, int kc) {
  const float4* xp = (const float4*)(xrow + kc * 32);
  float4 a0 = xp[0];
  float4 a1 = xp[1];
  float4 a2 = xp[4];   // +16 floats
  float4 a3 = xp[5];
  v16bf av;
  av[0]=(__bf16)a0.x;  av[1]=(__bf16)a0.y;  av[2]=(__bf16)a0.z;  av[3]=(__bf16)a0.w;
  av[4]=(__bf16)a1.x;  av[5]=(__bf16)a1.y;  av[6]=(__bf16)a1.z;  av[7]=(__bf16)a1.w;
  av[8]=(__bf16)a2.x;  av[9]=(__bf16)a2.y;  av[10]=(__bf16)a2.z; av[11]=(__bf16)a2.w;
  av[12]=(__bf16)a3.x; av[13]=(__bf16)a3.y; av[14]=(__bf16)a3.z; av[15]=(__bf16)a3.w;
  return av;
}

__global__ __launch_bounds__(kTPB)
void moe_gate_kernel(const float* __restrict__ x, const __bf16* __restrict__ wbf,
                     float* __restrict__ out_idx, float* __restrict__ out_w,
                     float* __restrict__ g_sums, float* __restrict__ g_cnts)
{
  __shared__ float lds_logits[kTokPB * kExp];   // 80 KB
  __shared__ float lds_psum[kExp];
  __shared__ float lds_cnt[kExp];

  const int tid  = threadIdx.x;
  const int lane = tid & 31;
  const int wave = tid >> 5;
  const int row  = lane & 15;   // A row / B column / C column
  const int hi   = lane >> 4;   // half-wave selector

  for (int i = tid; i < kExp; i += kTPB) { lds_psum[i] = 0.0f; lds_cnt[i] = 0.0f; }

  const int tokBase = blockIdx.x * kTokPB + wave * kTokPW;

  v8f acc0[kTiles], acc1[kTiles];
#pragma unroll
  for (int et = 0; et < kTiles; ++et) {
    acc0[et] = (v8f){0,0,0,0,0,0,0,0};
    acc1[et] = (v8f){0,0,0,0,0,0,0,0};
  }

  // A: lane holds row; halves 0..7 = k[hi*8 ..], halves 8..15 = k[16+hi*8 ..]
  const float*  xrow0 = x + (size_t)(tokBase + row) * kHid + hi * 8;
  const float*  xrow1 = xrow0 + (size_t)16 * kHid;
  // B: lane holds column (expert et*16+row); halves 0..7 = k[hi*8 ..], halves 8..15 = k[16+hi*8 ..]
  const __bf16* wcol  = wbf + (size_t)row * kHid + hi * 8;

  for (int kc = 0; kc < kChunks; ++kc) {
    v16bf av0 = cvt_a_frag(xrow0, kc);
    v16bf av1 = cvt_a_frag(xrow1, kc);

#pragma unroll
    for (int et = 0; et < kTiles; ++et) {
      const uint4* bp = (const uint4*)(wcol + (size_t)et * 16 * kHid + kc * 32);
      BV b;
      b.u[0] = bp[0];
      b.u[1] = bp[2];
      acc0[et] = __builtin_amdgcn_wmma_f32_16x16x32_bf16(
          false, av0, false, b.v, (short)0, acc0[et], false, false);
      acc1[et] = __builtin_amdgcn_wmma_f32_16x16x32_bf16(
          false, av1, false, b.v, (short)0, acc1[et], false, false);
    }
  }

  // C layout: lanes 0-15 -> M = vgpr (0..7); lanes 16-31 -> M = vgpr+8; N = lane&15
#pragma unroll
  for (int et = 0; et < kTiles; ++et) {
#pragma unroll
    for (int v = 0; v < 8; ++v) {
      lds_logits[(wave * kTokPW      + hi * 8 + v) * kExp + et * 16 + row] = acc0[et][v];
      lds_logits[(wave * kTokPW + 16 + hi * 8 + v) * kExp + et * 16 + row] = acc1[et][v];
    }
  }
  __syncthreads();

  {
    const int tt = tid;                         // one token per thread
    const int t  = blockIdx.x * kTokPB + tt;    // global token
    const float* L = &lds_logits[tt * kExp];

    // softmax statistics over logits
    float m = -3.4e38f;
    for (int e = 0; e < kExp; ++e) m = fmaxf(m, L[e]);
    float s = 0.0f;
    for (int e = 0; e < kExp; ++e) s += __expf(L[e] - m);
    const float inv = 1.0f / s;

    // Pi accumulation (mean of full softmax over tokens)
    for (int e = 0; e < kExp; ++e)
      atomicAdd(&lds_psum[e], __expf(L[e] - m) * inv);

    // group maxes on logits (monotone under softmax)
    float g[kGroups];
    for (int j = 0; j < kGroups; ++j) {
      float gm = -3.4e38f;
      for (int ii = 0; ii < kEpg; ++ii) gm = fmaxf(gm, L[j * kEpg + ii]);
      g[j] = gm;
    }
    // top-3 groups (strict > keeps lowest index on ties, like lax.top_k)
    unsigned gmask = 0;
    for (int k = 0; k < kTopG; ++k) {
      float best = -3.4e38f; int bj = 0;
      for (int j = 0; j < kGroups; ++j)
        if (!((gmask >> j) & 1u) && g[j] > best) { best = g[j]; bj = j; }
      gmask |= 1u << bj;
    }
    // top-6 experts within selected groups, index order preserved
    int isel[kTopK];
    for (int k = 0; k < kTopK; ++k) {
      float best = -3.4e38f; int bi = 0;
      for (int j = 0; j < kGroups; ++j) {
        if (!((gmask >> j) & 1u)) continue;
        for (int ii = 0; ii < kEpg; ++ii) {
          int e = j * kEpg + ii;
          bool used = false;
          for (int q = 0; q < k; ++q) used |= (isel[q] == e);
          if (!used && L[e] > best) { best = L[e]; bi = e; }
        }
      }
      isel[k] = bi;
    }

    for (int k = 0; k < kTopK; ++k) {
      out_idx[(size_t)t * kTopK + k] = (float)isel[k];
      out_w [(size_t)t * kTopK + k] = __expf(L[isel[k]] - m) * inv * 16.0f;
      atomicAdd(&lds_cnt[isel[k]], 1.0f);
    }
  }
  __syncthreads();

  for (int i = tid; i < kExp; i += kTPB) {
    atomicAdd(&g_sums[i], lds_psum[i]);
    atomicAdd(&g_cnts[i], lds_cnt[i]);
  }
}

__global__ void aux_kernel(const float* __restrict__ sums,
                           const float* __restrict__ cnts,
                           float* __restrict__ out)
{
  __shared__ float red[kExp];
  int i = threadIdx.x;
  if (i < kExp) {
    float Pi = sums[i] / (float)kTok;
    float fi = cnts[i] * (float)kExp / (float)(kTok * kTopK);
    red[i] = Pi * fi;
  }
  __syncthreads();
  if (i == 0) {
    float a = 0.0f;
    for (int e = 0; e < kExp; ++e) a += red[e];
    out[0] = a * 0.001f;
  }
}

extern "C" void kernel_launch(void* const* d_in, const int* in_sizes, int n_in,
                              void* d_out, int out_size, void* d_ws, size_t ws_size,
                              hipStream_t stream)
{
  (void)in_sizes; (void)n_in; (void)out_size; (void)ws_size;
  const float* x = (const float*)d_in[0];   // [4, 4096, 2048] f32
  const float* w = (const float*)d_in[1];   // [160, 2048] f32
  float* out = (float*)d_out;               // [T*6 idx | T*6 w | 1 aux]

  __bf16* wbf   = (__bf16*)d_ws;                                  // 640 KB bf16 weights
  float*  gsums = (float*)((char*)d_ws + (size_t)kExp * kHid * sizeof(__bf16));
  float*  gcnts = gsums + kExp;

  zero_accum<<<1, 2 * kExp, 0, stream>>>(gsums, 2 * kExp);

  const int nW = kExp * kHid;
  wconv<<<(nW + 255) / 256, 256, 0, stream>>>(w, wbf, nW);

  moe_gate_kernel<<<kTok / kTokPB, kTPB, 0, stream>>>(
      x, wbf, out, out + (size_t)kTok * kTopK, gsums, gcnts);

  aux_kernel<<<1, kExp, 0, stream>>>(gsums, gcnts, out + 2 * (size_t)kTok * kTopK);
}
